// ConRC_1013612282221
// MI455X (gfx1250) — compile-verified, matrix operand order checked by reading, not run on hardware
//
#include <hip/hip_runtime.h>
#include <hip/hip_bf16.h>
#include <math.h>

#define NN 50000
#define EE 600000
#define HH 128
#define TAU_ 0.5f

typedef __attribute__((ext_vector_type(16))) __bf16 v16bf;
typedef __attribute__((ext_vector_type(8)))  float v8f;

#if defined(__has_builtin)
#  if __has_builtin(__builtin_amdgcn_tensor_load_to_lds)
#    define USE_TDM 1
#  endif
#  if __has_builtin(__builtin_amdgcn_s_wait_tensorcnt)
#    define HAVE_TCNT 1
#  endif
#endif

#ifdef USE_TDM
typedef unsigned tdm_v4u __attribute__((ext_vector_type(4)));
typedef int      tdm_v8i __attribute__((ext_vector_type(8)));
typedef int      tdm_v4i __attribute__((ext_vector_type(4)));
#endif

// ---------------------------------------------------------------- utilities
__global__ void k_fill(float* p, long n, float v) {
  long i = (long)blockIdx.x * blockDim.x + threadIdx.x;
  if (i < n) p[i] = v;
}

__global__ void k_deg(const int* __restrict__ dst, float* deg, int e) {
  int i = blockIdx.x * blockDim.x + threadIdx.x;
  if (i < e) atomicAdd(&deg[dst[i]], 1.0f);
}

__global__ void k_hdeg(const int* __restrict__ node, const int* __restrict__ he,
                       float* Dn, float* Be, int e) {
  int i = blockIdx.x * blockDim.x + threadIdx.x;
  if (i < e) { atomicAdd(&Dn[node[i]], 1.0f); atomicAdd(&Be[he[i]], 1.0f); }
}

__global__ void k_invert(float* dinv, float* Binv, float* Dninv, int n) {
  int i = blockIdx.x * blockDim.x + threadIdx.x;
  if (i < n) {
    dinv[i] = rsqrtf(dinv[i]);                         // deg >= 1 (self loops)
    float b = Binv[i];  Binv[i]  = (b > 0.f) ? 1.f / b : 0.f;
    float d = Dninv[i]; Dninv[i] = (d > 0.f) ? 1.f / d : 0.f;
  }
}

// ------------------------------------------------ one-hot (querys) shortcuts
__global__ void k_qgcn(const int* __restrict__ src, const int* __restrict__ dst,
                       const int* qp, const float* dinv, float* t, int e) {
  int i = blockIdx.x * blockDim.x + threadIdx.x;
  if (i < e) {
    int q = *qp;
    if (src[i] == q) atomicAdd(&t[dst[i]], dinv[q] * dinv[dst[i]]);
  }
}
__global__ void k_selfq(float* t, const int* qp, const float* dinv) {
  if (threadIdx.x == 0 && blockIdx.x == 0) { int q = *qp; t[q] += dinv[q] * dinv[q]; }
}
__global__ void k_qhg1(const int* __restrict__ node, const int* __restrict__ he,
                       const int* qp, float* cq, int e) {
  int i = blockIdx.x * blockDim.x + threadIdx.x;
  if (i < e && node[i] == *qp) atomicAdd(&cq[he[i]], 1.0f);
}
__global__ void k_qhg2(const int* __restrict__ node, const int* __restrict__ he,
                       const float* Binv, const float* cq, float* sq, int e) {
  int i = blockIdx.x * blockDim.x + threadIdx.x;
  if (i < e) { int j = he[i]; float v = Binv[j] * cq[j]; if (v != 0.f) atomicAdd(&sq[node[i]], v); }
}

// ------------------------------------------- weight prep: fp32 -> bf16 hi/lo, transposed
__global__ void k_prepw(const float* __restrict__ W, unsigned short* hiT, unsigned short* loT) {
  int i = blockIdx.x * blockDim.x + threadIdx.x;   // 0..16383
  int k = i >> 7, n = i & 127;
  float w = W[k * 128 + n];
  unsigned u = __float_as_uint(w);
  unsigned short h = (unsigned short)(u >> 16);
  float fh = __uint_as_float(u & 0xffff0000u);
  unsigned short l = (unsigned short)(__float_as_uint(w - fh) >> 16);
  hiT[n * 128 + k] = h;
  loT[n * 128 + k] = l;
}

// ------------------------------------------------------------- WMMA GEMM
// Out(N x 128) = [relu?](A) (N x 128) @ W (128 x 128)  [+ bias] [relu on store]
// bf16x3 split: D += Ahi*Whi + Ahi*Wlo + Alo*Whi  (fp32 accumulate)
// B K-chunks staged into padded LDS rows by the Tensor Data Mover when available.
__global__ __launch_bounds__(256)
void k_gemm(const float* __restrict__ A, const unsigned short* __restrict__ BhiT,
            const unsigned short* __restrict__ BloT, const float* __restrict__ bias,
            float* __restrict__ Out, int nrows, int reluA, int reluOut) {
  __shared__ unsigned short lbh[128 * 40];   // K-chunk of W^T hi, 80B padded rows
  __shared__ unsigned short lbl[128 * 40];   // K-chunk of W^T lo
  const int tid = threadIdx.x, wave = tid >> 5, lane = tid & 31;
  const long strip = (long)blockIdx.x * 8 + wave;   // 16-row strip
  const long row16 = strip * 16;
  const int m = lane & 15, kh = lane >> 4;
  const bool active = row16 < (long)nrows;

  v8f acc[8];
#pragma unroll
  for (int t = 0; t < 8; t++)
#pragma unroll
    for (int j = 0; j < 8; j++) acc[t][j] = 0.f;

  union FragA { v16bf v; unsigned short u[16]; };
  union FragB { v16bf v; uint4 q[2]; };

#ifdef USE_TDM
  // D# group1 (constant across chunks):
  //   data_size=8B, pad_enable, pad_interval=16 DW, pad_amount=4 DW,
  //   tensor_dim0=8 units, tensor_dim1=128, tile_dim0=8 units, tile_dim1=128,
  //   tensor_dim0_stride=32 units (256B rows in global)
  const tdm_v8i g1v = {(int)0x06D30000, (int)0x00080000, (int)0x00800000,
                       (int)0x00080000, 128, 32, 0, 0};
  const tdm_v4i z4 = {0, 0, 0, 0};
#if __clang_major__ >= 23
  const tdm_v8i z8 = {0, 0, 0, 0, 0, 0, 0, 0};
#endif
#endif

  for (int kc = 0; kc < 4; ++kc) {
    __syncthreads();                                  // previous chunk consumed
#ifdef USE_TDM
    if (wave == 0) {
      unsigned ldsH = (unsigned)(unsigned long long)(const void*)lbh;
      unsigned ldsL = (unsigned)(unsigned long long)(const void*)lbl;
      unsigned long long gaH = (unsigned long long)(BhiT + kc * 32);
      unsigned long long gaL = (unsigned long long)(BloT + kc * 32);
      tdm_v4u g0H = {1u, ldsH, (unsigned)(gaH & 0xffffffffu),
                     (unsigned)((gaH >> 32) & 0x01ffffffu) | 0x80000000u};
      tdm_v4u g0L = {1u, ldsL, (unsigned)(gaL & 0xffffffffu),
                     (unsigned)((gaL >> 32) & 0x01ffffffu) | 0x80000000u};
#if __clang_major__ >= 23
      __builtin_amdgcn_tensor_load_to_lds(g0H, g1v, z4, z4, z8, 0);
      __builtin_amdgcn_tensor_load_to_lds(g0L, g1v, z4, z4, z8, 0);
#else
      __builtin_amdgcn_tensor_load_to_lds(g0H, g1v, z4, z4, 0);
      __builtin_amdgcn_tensor_load_to_lds(g0L, g1v, z4, z4, 0);
#endif
#ifdef HAVE_TCNT
      __builtin_amdgcn_s_wait_tensorcnt(0);
#else
      asm volatile("s_wait_tensorcnt 0x0" ::: "memory");
#endif
    }
#else
#pragma unroll
    for (int rep = 0; rep < 4; rep++) {               // stage 2*128*64B = 16KB
      int seg = rep * 256 + tid;
      int half = seg >> 9, rem = seg & 511, n = rem >> 2, part = rem & 3;
      const unsigned short* sp = (half ? BloT : BhiT) + n * 128 + kc * 32 + part * 8;
      unsigned short* dp = (half ? lbl : lbh) + n * 40 + part * 8;
      *(uint4*)dp = *(const uint4*)sp;
    }
#endif
    __syncthreads();
    if (active) {
      // A fragment: lane holds row (row16+m), K runs [kc*32+kh*8, +8) and [kc*32+16+kh*8, +8)
      const float* arow = A + (row16 + m) * 128 + kc * 32 + kh * 8;
      if (kc < 3) __builtin_prefetch(arow + 32, 0, 0);   // next K-chunk of A
      FragA ahi, alo;
#pragma unroll
      for (int h2 = 0; h2 < 2; h2++) {
        float4 f0 = *(const float4*)(arow + h2 * 16);
        float4 f1 = *(const float4*)(arow + h2 * 16 + 4);
        float fs[8] = {f0.x, f0.y, f0.z, f0.w, f1.x, f1.y, f1.z, f1.w};
#pragma unroll
        for (int j = 0; j < 8; j++) {
          float f = fs[j];
          if (reluA) f = fmaxf(f, 0.f);
          unsigned u = __float_as_uint(f);
          ahi.u[h2 * 8 + j] = (unsigned short)(u >> 16);
          float fh = __uint_as_float(u & 0xffff0000u);
          alo.u[h2 * 8 + j] = (unsigned short)(__float_as_uint(f - fh) >> 16);
        }
      }
#pragma unroll
      for (int t = 0; t < 8; t++) {
        const unsigned short* rb = lbh + (t * 16 + m) * 40 + kh * 8;
        const unsigned short* rl = lbl + (t * 16 + m) * 40 + kh * 8;
        FragB bh, bl;
        bh.q[0] = *(const uint4*)rb;          bh.q[1] = *(const uint4*)(rb + 16);
        bl.q[0] = *(const uint4*)rl;          bl.q[1] = *(const uint4*)(rl + 16);
        acc[t] = __builtin_amdgcn_wmma_f32_16x16x32_bf16(false, ahi.v, false, bh.v,
                                                         (short)0, acc[t], false, false);
        acc[t] = __builtin_amdgcn_wmma_f32_16x16x32_bf16(false, ahi.v, false, bl.v,
                                                         (short)0, acc[t], false, false);
        acc[t] = __builtin_amdgcn_wmma_f32_16x16x32_bf16(false, alo.v, false, bh.v,
                                                         (short)0, acc[t], false, false);
      }
    }
  }
  if (active) {
#pragma unroll
    for (int t = 0; t < 8; t++) {
      int col = t * 16 + m;
      float b = bias ? bias[col] : 0.f;
#pragma unroll
      for (int r = 0; r < 8; r++) {
        float v = acc[t][r] + b;
        if (reluOut) v = fmaxf(v, 0.f);
        Out[(row16 + r + 8 * kh) * 128 + col] = v;
      }
    }
  }
}

// -------------------------------------------- aggregation init + scatters
// out[i][:] = (accum? out[i][:] : 0) + bias + (xw? dinv[i]^2 * xw[i][:] : 0)
__global__ void k_rowinit(float* out, const float* __restrict__ bias,
                          const float* __restrict__ xw, const float* __restrict__ dinv,
                          int accum, int n) {
  long t = (long)blockIdx.x * blockDim.x + threadIdx.x;
  long i = t >> 5; int c = (int)(t & 31) * 4;
  if (i >= n) return;
  float4 b = *(const float4*)(bias + c);
  float4 o = {0.f, 0.f, 0.f, 0.f};
  if (accum) o = *(const float4*)(out + i * 128 + c);
  o.x += b.x; o.y += b.y; o.z += b.z; o.w += b.w;
  if (xw) {
    float w = dinv[i] * dinv[i];
    float4 x = *(const float4*)(xw + i * 128 + c);
    o.x += w * x.x; o.y += w * x.y; o.z += w * x.z; o.w += w * x.w;
  }
  *(float4*)(out + i * 128 + c) = o;
}

__global__ void k_gcnscatter(const int* __restrict__ src, const int* __restrict__ dst,
                             const float* __restrict__ dinv, const float* __restrict__ xw,
                             float* out, int e) {
  long t = (long)blockIdx.x * blockDim.x + threadIdx.x;
  long i = t >> 5; int c = (int)(t & 31) * 4;
  if (i >= e) return;
  int s = src[i], d = dst[i];
  float w = dinv[s] * dinv[d];
  float4 x = *(const float4*)(xw + (long)s * 128 + c);
  float* o = out + (long)d * 128 + c;
  atomicAdd(o + 0, w * x.x); atomicAdd(o + 1, w * x.y);
  atomicAdd(o + 2, w * x.z); atomicAdd(o + 3, w * x.w);
}

__global__ void k_hg1(const int* __restrict__ node, const int* __restrict__ he,
                      const float* __restrict__ xw, float* me, int e) {
  long t = (long)blockIdx.x * blockDim.x + threadIdx.x;
  long i = t >> 5; int c = (int)(t & 31) * 4;
  if (i >= e) return;
  int nd = node[i], j = he[i];
  float4 x = *(const float4*)(xw + (long)nd * 128 + c);
  float* o = me + (long)j * 128 + c;
  atomicAdd(o + 0, x.x); atomicAdd(o + 1, x.y); atomicAdd(o + 2, x.z); atomicAdd(o + 3, x.w);
}

__global__ void k_hg2(const int* __restrict__ node, const int* __restrict__ he,
                      const float* __restrict__ Binv, const float* __restrict__ Dninv,
                      const float* __restrict__ me, float* out, int e) {
  long t = (long)blockIdx.x * blockDim.x + threadIdx.x;
  long i = t >> 5; int c = (int)(t & 31) * 4;
  if (i >= e) return;
  int nd = node[i], j = he[i];
  float w = Dninv[nd] * Binv[j];
  float4 x = *(const float4*)(me + (long)j * 128 + c);
  float* o = out + (long)nd * 128 + c;
  atomicAdd(o + 0, w * x.x); atomicAdd(o + 1, w * x.y);
  atomicAdd(o + 2, w * x.z); atomicAdd(o + 3, w * x.w);
}

// ----------------------------------------- fuse: softmax-of-2 attention blend
// A-row = brow + cf*Wrow (cf = coef[i]*scale[i], or 1{i==q} if coef==null)
__global__ __launch_bounds__(256)
void k_fuse(const float* __restrict__ coef, const float* __restrict__ scale,
            const float* __restrict__ Wrow, const float* __restrict__ brow,
            const float* __restrict__ B, const float* __restrict__ wa,
            const float* __restrict__ wb, float* out, const int* qp,
            int reluA, int reluB, int n) {
  int wave = threadIdx.x >> 5, lane = threadIdx.x & 31;
  long i = (long)blockIdx.x * 8 + wave;
  if (i >= n) return;
  int q = *qp;
  float cf = coef ? (scale ? coef[i] * scale[i] : coef[i]) : ((i == q) ? 1.f : 0.f);
  int c = lane * 4;
  float4 br = *(const float4*)(brow + c);
  float4 wr = *(const float4*)(Wrow + c);
  float4 bm = *(const float4*)(B + i * 128 + c);
  float4 va = *(const float4*)(wa + c);
  float4 vb = *(const float4*)(wb + c);
  float a0 = br.x + cf * wr.x, a1 = br.y + cf * wr.y, a2 = br.z + cf * wr.z, a3 = br.w + cf * wr.w;
  if (reluA) { a0 = fmaxf(a0, 0.f); a1 = fmaxf(a1, 0.f); a2 = fmaxf(a2, 0.f); a3 = fmaxf(a3, 0.f); }
  float b0 = bm.x, b1 = bm.y, b2 = bm.z, b3 = bm.w;
  if (reluB) { b0 = fmaxf(b0, 0.f); b1 = fmaxf(b1, 0.f); b2 = fmaxf(b2, 0.f); b3 = fmaxf(b3, 0.f); }
  float la = a0 * va.x + a1 * va.y + a2 * va.z + a3 * va.w;
  float lb = b0 * vb.x + b1 * vb.y + b2 * vb.z + b3 * vb.w;
  for (int msk = 16; msk > 0; msk >>= 1) {
    la += __shfl_xor(la, msk, 32);
    lb += __shfl_xor(lb, msk, 32);
  }
  float att = 1.f / (1.f + expf(lb - la));   // softmax([la,lb])[0]
  float4 o;
  o.x = att * a0 + (1.f - att) * b0; o.y = att * a1 + (1.f - att) * b1;
  o.z = att * a2 + (1.f - att) * b2; o.w = att * a3 + (1.f - att) * b3;
  *(float4*)(out + i * 128 + c) = o;
}

// ---------------------------------------------------------- loss kernels
__global__ __launch_bounds__(256)
void k_rownorm(const float* __restrict__ M, float* nrm, int n) {
  int wave = threadIdx.x >> 5, lane = threadIdx.x & 31;
  long i = (long)blockIdx.x * 8 + wave;
  if (i >= n) return;
  float4 x = *(const float4*)(M + i * 128 + lane * 4);
  float s = x.x * x.x + x.y * x.y + x.z * x.z + x.w * x.w;
  for (int msk = 16; msk > 0; msk >>= 1) s += __shfl_xor(s, msk, 32);
  if (lane == 0) nrm[i] = sqrtf(s);
}

__global__ void k_setmask(const int* __restrict__ pos, float* mask, int np) {
  int i = blockIdx.x * blockDim.x + threadIdx.x;
  if (i < np) mask[pos[i]] = 1.0f;
}
__global__ void k_clearq(float* mask, const int* qp) {
  if (threadIdx.x == 0 && blockIdx.x == 0) mask[*qp] = 0.0f;
}

__global__ __launch_bounds__(256)
void k_sim(const float* __restrict__ z, const float* __restrict__ za,
           const float* __restrict__ nz, const float* __restrict__ nza,
           const float* __restrict__ mask, const int* qp, float* S, int n) {
  int wave = threadIdx.x >> 5, lane = threadIdx.x & 31;
  long i = (long)blockIdx.x * 8 + wave;
  if (i >= n) return;
  int q = *qp;
  int c = lane * 4;
  float4 zi = *(const float4*)(z + i * 128 + c);
  float4 zai = *(const float4*)(za + i * 128 + c);
  float4 v1 = *(const float4*)(z + (long)q * 128 + c);
  float4 v2 = *(const float4*)(za + (long)q * 128 + c);
  float d11 = zi.x * v1.x + zi.y * v1.y + zi.z * v1.z + zi.w * v1.w;
  float d12 = zi.x * v2.x + zi.y * v2.y + zi.z * v2.z + zi.w * v2.w;
  float d21 = zai.x * v1.x + zai.y * v1.y + zai.z * v1.z + zai.w * v1.w;
  float d22 = zai.x * v2.x + zai.y * v2.y + zai.z * v2.z + zai.w * v2.w;
  for (int msk = 16; msk > 0; msk >>= 1) {
    d11 += __shfl_xor(d11, msk, 32); d12 += __shfl_xor(d12, msk, 32);
    d21 += __shfl_xor(d21, msk, 32); d22 += __shfl_xor(d22, msk, 32);
  }
  if (lane == 0) {
    float nzq = nz[q], nzaq = nza[q], nzi = nz[i], nzai = nza[i];
    float c1 = d11 / fmaxf(nzi * nzq, 1e-8f) / TAU_;    // sim1
    float c2 = d22 / fmaxf(nzai * nzaq, 1e-8f) / TAU_;  // sim_aug1
    float c3 = d21 / fmaxf(nzai * nzq, 1e-8f) / TAU_;   // sim2
    float c4 = d12 / fmaxf(nzi * nzaq, 1e-8f) / TAU_;   // sim_aug2
    atomicAdd(&S[0], expf(c1)); atomicAdd(&S[1], expf(c2));
    atomicAdd(&S[2], expf(c3)); atomicAdd(&S[3], expf(c4));
    float mk = mask[i];
    if (mk != 0.f) {
      atomicAdd(&S[4], mk * c1); atomicAdd(&S[5], mk * c2);
      atomicAdd(&S[6], mk * c3); atomicAdd(&S[7], mk * c4);
      atomicAdd(&S[8], mk);
    }
  }
}

__global__ void k_final(const float* __restrict__ z, const float* __restrict__ za,
                        const float* __restrict__ nz, const float* __restrict__ nza,
                        const int* qp, const float* S, float* out) {
  int lane = threadIdx.x & 31;
  int q = *qp;
  int c = lane * 4;
  float4 a = *(const float4*)(z + (long)q * 128 + c);
  float4 b = *(const float4*)(za + (long)q * 128 + c);
  float d = a.x * b.x + a.y * b.y + a.z * b.z + a.w * b.w;
  for (int msk = 16; msk > 0; msk >>= 1) d += __shfl_xor(d, msk, 32);
  if (lane == 0) {
    float csqq = d / fmaxf(nz[q] * nza[q], 1e-8f) / TAU_;
    float S1 = S[0], S2 = S[1], S3 = S[2], S4 = S[3];
    float L1 = S[4], L2 = S[5], L3 = S[6], L4 = S[7], nm = S[8];
    float nlm1 = logf(S1) - L1 / nm, nlm2 = logf(S2) - L2 / nm;
    float nlm3 = logf(S3) - L3 / nm, nlm4 = logf(S4) - L4 / nm;
    float loss_intra = 0.5f * (nlm1 + nlm2);
    float loss_inter = 0.5f * (nlm3 + nlm4);
    float loss_unsup = 0.5f * (logf(S3) - csqq) + 0.5f * (logf(S4) - csqq);
    out[0] = loss_intra + 0.5f * loss_inter + 0.5f * loss_unsup;
  }
}

// ================================================================ host side
extern "C" void kernel_launch(void* const* d_in, const int* in_sizes, int n_in,
                              void* d_out, int out_size, void* d_ws, size_t ws_size,
                              hipStream_t stream) {
  (void)in_sizes; (void)n_in; (void)out_size; (void)ws_size;
  const float* feats = (const float*)d_in[0];
  const int*   edge  = (const int*)d_in[1];   // [0..E)=src, [E..2E)=dst
  const int*   aug   = (const int*)d_in[2];   // [0..E)=node, [E..2E)=he
  const int*   pos   = (const int*)d_in[3];
  const int*   qp    = (const int*)d_in[4];
  const float* Wq0 = (const float*)d_in[5],  *bq0 = (const float*)d_in[6];
  const float* W0  = (const float*)d_in[7],  *b0  = (const float*)d_in[8];
  const float* Whq0= (const float*)d_in[9],  *bhq0= (const float*)d_in[10];
  const float* Wh0 = (const float*)d_in[11], *bh0 = (const float*)d_in[12];
  const float* Wf0 = (const float*)d_in[13], *bf0 = (const float*)d_in[14];
  const float* Wfh0= (const float*)d_in[15], *bfh0= (const float*)d_in[16];
  const float* aq0 = (const float*)d_in[17], *a0  = (const float*)d_in[18];
  const float* ahq0= (const float*)d_in[19], *ah0 = (const float*)d_in[20];
  const float* aq_ = (const float*)d_in[21], *a_  = (const float*)d_in[22];
  const float* ahq_= (const float*)d_in[23], *ah_ = (const float*)d_in[24];
  const float* Wlq = (const float*)d_in[25], *blq = (const float*)d_in[26];
  const float* Wlf = (const float*)d_in[27], *blf = (const float*)d_in[28];
  const float* Wm1 = (const float*)d_in[29], *bm1 = (const float*)d_in[30];
  const float* Wm2 = (const float*)d_in[31], *bm2 = (const float*)d_in[32];

  // ---- workspace layout
  char* ws = (char*)d_ws;
  size_t off = 0;
  auto alloc = [&](size_t bytes) -> void* {
    void* p = ws + off; off += (bytes + 255) & ~(size_t)255; return p;
  };
  float* bufA = (float*)alloc((size_t)NN * HH * 4);
  float* bufB = (float*)alloc((size_t)NN * HH * 4);
  float* bufC = (float*)alloc((size_t)NN * HH * 4);
  float* bufD = (float*)alloc((size_t)NN * HH * 4);
  float* bufE = (float*)alloc((size_t)NN * HH * 4);
  float* bufF = (float*)alloc((size_t)NN * HH * 4);
  float* dinv  = (float*)alloc(NN * 4);
  float* Binv  = (float*)alloc(NN * 4);
  float* Dninv = (float*)alloc(NN * 4);
  float* tq    = (float*)alloc(NN * 4);
  float* sq    = (float*)alloc(NN * 4);
  float* cq    = (float*)alloc(NN * 4);
  float* maskv = (float*)alloc(NN * 4);
  float* nzv   = (float*)alloc(NN * 4);
  float* nzav  = (float*)alloc(NN * 4);
  float* Sacc  = (float*)alloc(64 * 4);
  unsigned short* wbf = (unsigned short*)alloc((size_t)7 * 2 * 16384 * 2);

  auto g1 = [](long n) { return dim3((unsigned)((n + 255) / 256)); };
  const dim3 B256(256);
  const dim3 ROWG((NN + 7) / 8);          // wave-per-row kernels
  const long NH = (long)NN * HH;
  const long N32 = (long)NN * 32;
  const long E32 = (long)EE * 32;

  // ---- degrees / inverses
  k_fill<<<g1(NN), B256, 0, stream>>>(dinv, NN, 1.0f);          // self loops
  k_fill<<<g1(NN), B256, 0, stream>>>(Binv, NN, 0.0f);
  k_fill<<<g1(NN), B256, 0, stream>>>(Dninv, NN, 0.0f);
  k_deg<<<g1(EE), B256, 0, stream>>>(edge + EE, dinv, EE);
  k_hdeg<<<g1(EE), B256, 0, stream>>>(aug, aug + EE, Dninv, Binv, EE);
  k_invert<<<g1(NN), B256, 0, stream>>>(dinv, Binv, Dninv, NN);

  // ---- one-hot query coefficients
  k_fill<<<g1(NN), B256, 0, stream>>>(tq, NN, 0.0f);
  k_fill<<<g1(NN), B256, 0, stream>>>(sq, NN, 0.0f);
  k_fill<<<g1(NN), B256, 0, stream>>>(cq, NN, 0.0f);
  k_fill<<<g1(NN), B256, 0, stream>>>(maskv, NN, 0.0f);
  k_fill<<<1, B256, 0, stream>>>(Sacc, 64, 0.0f);
  k_qgcn<<<g1(EE), B256, 0, stream>>>(edge, edge + EE, qp, dinv, tq, EE);
  k_selfq<<<1, 1, 0, stream>>>(tq, qp, dinv);
  k_qhg1<<<g1(EE), B256, 0, stream>>>(aug, aug + EE, qp, cq, EE);
  k_qhg2<<<g1(EE), B256, 0, stream>>>(aug, aug + EE, Binv, cq, sq, EE);

  // ---- weight prep (bf16 hi/lo, transposed): slots
  auto wHi = [&](int s) { return wbf + (size_t)s * 32768; };
  auto wLo = [&](int s) { return wbf + (size_t)s * 32768 + 16384; };
  const float* wsrc[7] = {W0, Wh0, Wlf, Wf0, Wfh0, Wm1, Wm2};
  for (int s = 0; s < 7; s++)
    k_prepw<<<64, B256, 0, stream>>>(wsrc[s], wHi(s), wLo(s));

  const dim3 GEMMG((NN / 16 + 7) / 8);   // 391 blocks of 8 strips
  auto gemm = [&](const float* Ain, int slot, const float* bias, float* Outp,
                  int rA, int rO) {
    k_gemm<<<GEMMG, B256, 0, stream>>>(Ain, wHi(slot), wLo(slot), bias, Outp, NN, rA, rO);
  };

  // ---- h = gcn(feats, W0, b0)   (pre-relu in bufB)
  gemm(feats, 0, nullptr, bufA, 0, 0);
  k_rowinit<<<g1(N32), B256, 0, stream>>>(bufB, b0, bufA, dinv, 0, NN);
  k_gcnscatter<<<g1(E32), B256, 0, stream>>>(edge, edge + EE, dinv, bufA, bufB, EE);

  // ---- h_aug = hg(feats, Wh0, bh0)   (pre-relu in bufD)
  gemm(feats, 1, nullptr, bufA, 0, 0);
  k_fill<<<g1(NH), B256, 0, stream>>>(bufC, NH, 0.0f);
  k_hg1<<<g1(E32), B256, 0, stream>>>(aug, aug + EE, bufA, bufC, EE);
  k_rowinit<<<g1(N32), B256, 0, stream>>>(bufD, bh0, nullptr, nullptr, 0, NN);
  k_hg2<<<g1(E32), B256, 0, stream>>>(aug, aug + EE, Binv, Dninv, bufC, bufD, EE);

  // ---- hf = fuse(relu(hq), relu(h))  -> bufE ;  h_augf -> bufF
  k_fuse<<<ROWG, B256, 0, stream>>>(tq, nullptr, Wq0, bq0, bufB, aq0, a0, bufE, qp, 1, 1, NN);
  k_fuse<<<ROWG, B256, 0, stream>>>(sq, Dninv, Whq0, bhq0, bufD, ahq0, ah0, bufF, qp, 1, 1, NN);

  // ---- feats2 = feats @ Wlf + blf -> bufB
  gemm(feats, 2, blf, bufB, 0, 0);

  // ---- hf_ = fuse(querys2, feats2; aq_, a_) -> bufD ; gcn(hf_, Wf0, bf0) added into bufE
  k_fuse<<<ROWG, B256, 0, stream>>>(nullptr, nullptr, Wlq, blq, bufB, aq_, a_, bufD, qp, 0, 0, NN);
  gemm(bufD, 3, nullptr, bufA, 0, 0);
  k_rowinit<<<g1(N32), B256, 0, stream>>>(bufE, bf0, bufA, dinv, 1, NN);
  k_gcnscatter<<<g1(E32), B256, 0, stream>>>(edge, edge + EE, dinv, bufA, bufE, EE);

  // ---- hfh_ = fuse(querys2, feats2; ahq_, ah_) -> bufD ; hg(hfh_, Wfh0, bfh0) into bufF
  k_fuse<<<ROWG, B256, 0, stream>>>(nullptr, nullptr, Wlq, blq, bufB, ahq_, ah_, bufD, qp, 0, 0, NN);
  gemm(bufD, 4, nullptr, bufA, 0, 0);
  k_fill<<<g1(NH), B256, 0, stream>>>(bufC, NH, 0.0f);
  k_hg1<<<g1(E32), B256, 0, stream>>>(aug, aug + EE, bufA, bufC, EE);
  k_rowinit<<<g1(N32), B256, 0, stream>>>(bufF, bfh0, nullptr, nullptr, 1, NN);
  k_hg2<<<g1(E32), B256, 0, stream>>>(aug, aug + EE, Binv, Dninv, bufC, bufF, EE);

  // ---- MLP: z = relu(relu(hf)@Wm1+bm1)@Wm2+bm2 ; same for z_aug
  gemm(bufE, 5, bm1, bufA, 1, 1);     // relu(hf) on load, relu on store
  gemm(bufA, 6, bm2, bufB, 0, 0);     // z   = bufB
  gemm(bufF, 5, bm1, bufA, 1, 1);
  gemm(bufA, 6, bm2, bufC, 0, 0);     // z_aug = bufC

  // ---- loss
  k_rownorm<<<ROWG, B256, 0, stream>>>(bufB, nzv, NN);
  k_rownorm<<<ROWG, B256, 0, stream>>>(bufC, nzav, NN);
  k_setmask<<<1, 128, 0, stream>>>(pos, maskv, 100);
  k_clearq<<<1, 1, 0, stream>>>(maskv, qp);
  k_sim<<<ROWG, B256, 0, stream>>>(bufB, bufC, nzv, nzav, maskv, qp, Sacc, NN);
  k_final<<<1, 32, 0, stream>>>(bufB, bufC, nzv, nzav, qp, Sacc, (float*)d_out);
}
